// MultiHeadAttentionBlock_69106023792731
// MI455X (gfx1250) — matrix operand order repaired and checked by reading, not compiled, not run on hardware
//
#include <hip/hip_runtime.h>
#include <cstdint>
#include <cstddef>

// ---------------------------------------------------------------------------
// Problem constants (match reference)
// ---------------------------------------------------------------------------
#define D_MODEL 1024
#define NHEAD   16
#define DKH     64
#define BATCH   2
#define SEQ     2048

typedef __bf16 bf16_t;
typedef __attribute__((ext_vector_type(16))) bf16_t        v16bf;
typedef __attribute__((ext_vector_type(8)))  float         v8f;
typedef __attribute__((ext_vector_type(4)))  unsigned int  v4u;

// ---------------------------------------------------------------------------
// CDNA5 async global->LDS copy (ASYNCcnt-tracked), with guarded fallback
// ---------------------------------------------------------------------------
#if defined(__has_builtin)
#  if __has_builtin(__builtin_amdgcn_global_load_async_to_lds_b128)
#    define HAVE_ASYNC_COPY 1
#  endif
#  if __has_builtin(__builtin_amdgcn_s_wait_asynccnt)
#    define HAVE_WAIT_ASYNC 1
#  endif
#endif
#ifndef HAVE_ASYNC_COPY
#  define HAVE_ASYNC_COPY 0
#endif
#ifndef HAVE_WAIT_ASYNC
#  define HAVE_WAIT_ASYNC 0
#endif

// Exact pointee type the builtin expects: <4 x i32>
typedef __attribute__((__vector_size__(4 * sizeof(int)))) int v4i_t;
typedef __attribute__((address_space(1))) v4i_t g_v4i;   // global
typedef __attribute__((address_space(3))) v4i_t l_v4i;   // LDS

// Copies 16 bytes per lane: LDS[l] = MEM[g].
__device__ inline void async_copy_b128(const void* g, void* l) {
#if HAVE_ASYNC_COPY
    __builtin_amdgcn_global_load_async_to_lds_b128(
        (g_v4i*)(g), (l_v4i*)(l), /*offset=*/0, /*cpol=*/0);
#else
    *(v4u*)l = *(const v4u*)g;
#endif
}

__device__ inline void wait_async() {
#if HAVE_ASYNC_COPY
#  if HAVE_WAIT_ASYNC
    __builtin_amdgcn_s_wait_asynccnt(0);
#  else
    asm volatile("s_wait_asynccnt 0x0" ::: "memory");
#  endif
#endif
}

union ABFrag {
    struct { v4u lo, hi; } u;   // 2 x 16B = 32B
    v16bf v;                    // 16 x bf16 = 32B
};

// f32 -> bf16 round-to-nearest-even
__device__ inline bf16_t f2bf(float f) {
    unsigned u = __builtin_bit_cast(unsigned, f);
    u += 0x7FFFu + ((u >> 16) & 1u);
    unsigned short s = (unsigned short)(u >> 16);
    return __builtin_bit_cast(bf16_t, s);
}

// ---------------------------------------------------------------------------
// WMMA fragment loaders (CDNA5 ISA layouts, wave32)
// A 16x32 bf16: lane l holds row (l&15); k-base = 8*(l>>4); elems 0..7 at
//   k=kb..kb+7, elems 8..15 at k=16+kb..16+kb+7.
// B 32x16 bf16 staged as [N][K] in LDS: lane l holds col (l&15);
//   same k order as A: k-base = 8*(l>>4); elems 0..7 at k=kb..kb+7, 8..15 at k=16+kb..16+kb+7.
// C 16x16 f32: VGPR r -> row r + 8*(l>>4), col (l&15).
// ld (in bf16 elems) must keep 16-byte alignment: ld*2 % 16 == 0.
// ---------------------------------------------------------------------------
__device__ inline v16bf load_a_frag(const bf16_t* base, int ld, int lane) {
    const int row = lane & 15;
    const int kb  = (lane >> 4) * 8;
    ABFrag f;
    f.u.lo = *(const v4u*)(base + row * ld + kb);
    f.u.hi = *(const v4u*)(base + row * ld + kb + 16);
    return f.v;
}

__device__ inline v16bf load_b_frag(const bf16_t* base, int ld, int lane) {
    const int n  = lane & 15;
    const int kb = (lane >> 4) * 8;
    ABFrag f;
    f.u.lo = *(const v4u*)(base + n * ld + kb);
    f.u.hi = *(const v4u*)(base + n * ld + kb + 16);
    return f.v;
}

__device__ inline v8f wmma_bf16(v16bf a, v16bf b, v8f c) {
    // D = A x B + C, f32 accumulate
    return __builtin_amdgcn_wmma_f32_16x16x32_bf16(
        /*neg_a=*/false, a, /*neg_b=*/false, b,
        /*c_mod=*/(short)0, c, /*reuse_a=*/false, /*reuse_b=*/false);
}

// ---------------------------------------------------------------------------
// f32 -> bf16 conversion kernels (plain and transposing)
// ---------------------------------------------------------------------------
__global__ void cvt_f32_bf16(const float* __restrict__ src,
                             bf16_t* __restrict__ dst, int n) {
    int i = blockIdx.x * 256 + threadIdx.x;
    if (i < n) dst[i] = f2bf(src[i]);
}

// dst[n*rows + k] = src[k*cols + n]   (weights [K][N] -> [N][K] bf16)
__global__ void cvt_f32_bf16_T(const float* __restrict__ src,
                               bf16_t* __restrict__ dst, int rows, int cols) {
    int i = blockIdx.x * 256 + threadIdx.x;
    if (i < rows * cols) {
        int n = i / rows, k = i - n * rows;
        dst[i] = f2bf(src[(size_t)k * cols + n]);
    }
}

// ---------------------------------------------------------------------------
// Generic bf16 GEMM:  out = A[MxK] * Bt^T + bias  where Bt is [N][K] row-major
// 128 threads = 4 waves, workgroup tile 64x64, K-step 32, double-buffered
// async staging. Output modes: outF (f32 row-major), outB (bf16 row-major),
// outT (bf16 head-transposed: [b][h][d][s], for the V projection).
// ---------------------------------------------------------------------------
#define G_LDA 40   // 32 + 8 pad (bf16 units); 40*2 = 80B, 16B-aligned rows
#define G_LDB 40

__global__ __launch_bounds__(128)
void gemm_bf16(const bf16_t* __restrict__ A, const bf16_t* __restrict__ Bt,
               const float* __restrict__ bias,
               float* __restrict__ outF, bf16_t* __restrict__ outB,
               bf16_t* __restrict__ outT,
               int M, int N, int K) {
    __shared__ __align__(16) bf16_t As[2 * 64 * G_LDA];
    __shared__ __align__(16) bf16_t Bs[2 * 64 * G_LDB];

    const int tid  = threadIdx.x;
    const int lane = tid & 31;
    const int wave = tid >> 5;
    const int wr   = wave >> 1;     // wave row (0..1) -> 32 rows each
    const int wc   = wave & 1;      // wave col (0..1) -> 32 cols each
    const int hi   = lane >> 4;
    const int ln   = lane & 15;

    const int m0 = blockIdx.y * 64;
    const int n0 = blockIdx.x * 64;

    // Fully-async tile staging (no VGPR data movement).
    auto stage = [&](int k0, int buf) {
        bf16_t* as = As + buf * 64 * G_LDA;
        bf16_t* bs = Bs + buf * 64 * G_LDB;
        // A tile 64x16 (k0..k0+15 only): 64 rows x 2 chunks of 16B
        for (int c = tid; c < 64 * 2; c += 128) {
            int r = c >> 1, p = c & 1;
            async_copy_b128(&A[(size_t)(m0 + r) * K + k0 + p * 8],
                            &as[r * G_LDA + p * 8]);
        }
        // B tile (from [N][K] weights) 64x16 (k0..k0+15 only): 64 rows x 2 chunks of 16B
        for (int c = tid; c < 64 * 2; c += 128) {
            int n = c >> 1, p = c & 1;
            async_copy_b128(&Bt[(size_t)(n0 + n) * K + k0 + p * 8],
                            &bs[n * G_LDB + p * 8]);
        }
    };

    v8f acc[2][2] = {};   // [mi][ni] 16x16 tiles

    stage(0, 0);
    wait_async();
    __syncthreads();

    for (int k0 = 0; k0 < K; k0 += 32) {
        const int buf = (k0 >> 5) & 1;
        // Stage next tile into the other buffer while computing this one.
        if (k0 + 32 < K) stage(k0 + 32, buf ^ 1);
        // Prefetch two tiles ahead (global_prefetch_b8)
        if (k0 + 64 < K) {
            __builtin_prefetch(&A[(size_t)(m0 + (tid >> 1)) * K + k0 + 64 +
                                  (tid & 1) * 16], 0, 0);
            __builtin_prefetch(&Bt[(size_t)(n0 + (tid >> 1)) * K + k0 + 64 +
                                   (tid & 1) * 16], 0, 0);
        }

        const bf16_t* as = As + buf * 64 * G_LDA;
        const bf16_t* bs = Bs + buf * 64 * G_LDB;
        v16bf afr[2], bfr[2];
#pragma unroll
        for (int mi = 0; mi < 2; ++mi)
            afr[mi] = load_a_frag(as + (wr * 32 + mi * 16) * G_LDA, G_LDA, lane);
#pragma unroll
        for (int ni = 0; ni < 2; ++ni)
            bfr[ni] = load_b_frag(bs + (wc * 32 + ni * 16) * G_LDB, G_LDB, lane);
#pragma unroll
        for (int mi = 0; mi < 2; ++mi)
#pragma unroll
            for (int ni = 0; ni < 2; ++ni)
                acc[mi][ni] = wmma_bf16(afr[mi], bfr[ni], acc[mi][ni]);

        wait_async();      // next tile landed in LDS
        __syncthreads();   // everyone done reading current tile
    }

    // Epilogue: add bias, write out
#pragma unroll
    for (int mi = 0; mi < 2; ++mi) {
#pragma unroll
        for (int ni = 0; ni < 2; ++ni) {
            int col = n0 + wc * 32 + ni * 16 + ln;
            float bv = bias ? bias[col] : 0.0f;
            int rowbase = m0 + wr * 32 + mi * 16 + hi * 8;
            if (outF) {
#pragma unroll
                for (int r = 0; r < 8; ++r)
                    outF[(size_t)(rowbase + r) * N + col] = acc[mi][ni][r] + bv;
            } else if (outT) {
                // Head-transposed: addr = ((b*NHEAD + h)*DKH + d)*SEQ + s.
                // The 8 C rows per lane are consecutive s -> one 16B store.
                int bb = rowbase >> 11;        // / SEQ
                int s0 = rowbase & (SEQ - 1);
                int hh = col >> 6, dd = col & 63;
                union { bf16_t h[8]; v4u v; } tmp;
#pragma unroll
                for (int r = 0; r < 8; ++r) tmp.h[r] = f2bf(acc[mi][ni][r] + bv);
                *(v4u*)&outT[((size_t)(bb * NHEAD + hh) * DKH + dd) * SEQ + s0]
                    = tmp.v;
            } else {
#pragma unroll
                for (int r = 0; r < 8; ++r)
                    outB[(size_t)(rowbase + r) * N + col]
                        = f2bf(acc[mi][ni][r] + bv);
            }
        }
    }
}

// ---------------------------------------------------------------------------
// Flash-attention kernel, one workgroup per (batch, head, 64-query tile).
// 128 threads = 4 waves; wave w owns query rows w*16..w*16+15.
// V comes pre-transposed ([b][h][d][s]) so ALL tile staging is async.
// ---------------------------------------------------------------------------
#define A_LD 72   // 64 + 8 pad; 72*2 = 144B, 16B-aligned rows
#define M_LD 68   // mask stride in ints; 68*4 = 272B, 16B-aligned rows

__global__ __launch_bounds__(128)
void attention_bf16(const bf16_t* __restrict__ Qp, const bf16_t* __restrict__ Kp,
                    const bf16_t* __restrict__ VpT, const int* __restrict__ mask,
                    bf16_t* __restrict__ AO) {
    __shared__ __align__(16) bf16_t Qs[64 * A_LD];  // [q][d]
    __shared__ __align__(16) bf16_t Ks[64 * A_LD];  // [key][d]  (B-layout QK^T)
    __shared__ __align__(16) bf16_t Vs[64 * A_LD];  // [d][key]  (B-layout P*V)
    __shared__ __align__(16) bf16_t Ps[64 * A_LD];  // [q][key]  (A-layout P*V)
    __shared__ __align__(16) int    Msk[64 * M_LD]; // [q][key] raw int32

    const int tid  = threadIdx.x;
    const int lane = tid & 31;
    const int wave = tid >> 5;
    const int hi   = lane >> 4;
    const int ln   = lane & 15;

    const int qt = blockIdx.x;
    const int h  = blockIdx.y;
    const int b  = blockIdx.z;
    const int q0 = qt * 64;

    const bf16_t* Qg  = Qp  + ((size_t)b * SEQ + q0) * D_MODEL + h * DKH;
    const bf16_t* VTg = VpT + (size_t)(b * NHEAD + h) * DKH * SEQ;

    // Stage Q tile (64x64 bf16 = 64 rows x 8 chunks of 16B) via async
    for (int c = tid; c < 64 * 8; c += 128) {
        int r = c >> 3, p = c & 7;
        async_copy_b128(&Qg[(size_t)r * D_MODEL + p * 8], &Qs[r * A_LD + p * 8]);
    }

    float Mrun[8], Lrun[8];
#pragma unroll
    for (int r = 0; r < 8; ++r) { Mrun[r] = -3.0e38f; Lrun[r] = 0.0f; }
    v8f O[4] = {};

    for (int kt = 0; kt < SEQ / 64; ++kt) {
        const int k0 = kt * 64;
        __syncthreads();   // previous iteration's reads of Ks/Vs/Ps done

        const bf16_t* Kg = Kp + ((size_t)b * SEQ + k0) * D_MODEL + h * DKH;
        // K tile: direct copy via async (64 rows x 8 chunks)
        for (int c = tid; c < 64 * 8; c += 128) {
            int kk = c >> 3, p = c & 7;
            async_copy_b128(&Kg[(size_t)kk * D_MODEL + p * 8],
                            &Ks[kk * A_LD + p * 8]);
        }
        // V tile: pre-transposed source -> direct async copy [d][kk]
        for (int c = tid; c < 64 * 8; c += 128) {
            int d = c >> 3, p = c & 7;
            async_copy_b128(&VTg[(size_t)d * SEQ + k0 + p * 8],
                            &Vs[d * A_LD + p * 8]);
        }
        // Mask tile: raw int rows via async (64 rows x 16 chunks of 4 ints)
        const int* Mg = mask + ((size_t)b * SEQ + q0) * SEQ + k0;
        for (int c = tid; c < 64 * 16; c += 128) {
            int qq = c >> 4, p = c & 15;
            async_copy_b128(&Mg[(size_t)qq * SEQ + p * 4],
                            &Msk[qq * M_LD + p * 4]);
        }
        wait_async();
        __syncthreads();

        // ---- scores = Q * K^T (16x64 per wave), K-dim = 64 -> 2 WMMA steps
        v8f sc[4] = {};
#pragma unroll
        for (int ks = 0; ks < 2; ++ks) {
            v16bf aq = load_a_frag(Qs + wave * 16 * A_LD + ks * 32, A_LD, lane);
#pragma unroll
            for (int f = 0; f < 4; ++f) {
                v16bf bk = load_b_frag(Ks + f * 16 * A_LD + ks * 32, A_LD, lane);
                sc[f] = wmma_bf16(aq, bk, sc[f]);
            }
        }

        // ---- scale + mask, local row max
        float lmax[8];
#pragma unroll
        for (int r = 0; r < 8; ++r) lmax[r] = -3.0e38f;
#pragma unroll
        for (int f = 0; f < 4; ++f) {
#pragma unroll
            for (int r = 0; r < 8; ++r) {
                int qq = wave * 16 + hi * 8 + r;
                int kk = f * 16 + ln;
                float s = sc[f][r] * 0.125f;                 // 1/sqrt(64)
                if (Msk[qq * M_LD + kk] == 0) s = -1.0e9f;
                sc[f][r] = s;
                lmax[r] = fmaxf(lmax[r], s);
            }
        }
        // reduce max over the 16 lanes of each half (rows stay in-half)
#pragma unroll
        for (int off = 8; off >= 1; off >>= 1)
#pragma unroll
            for (int r = 0; r < 8; ++r)
                lmax[r] = fmaxf(lmax[r], __shfl_xor(lmax[r], off, 32));

        float alpha[8], lsum[8];
#pragma unroll
        for (int r = 0; r < 8; ++r) {
            float mnew = fmaxf(Mrun[r], lmax[r]);
            alpha[r] = __expf(Mrun[r] - mnew);
            Mrun[r]  = mnew;
            lsum[r]  = 0.0f;
        }
        // ---- exponentials -> P tile in LDS (A-layout), row sums
#pragma unroll
        for (int f = 0; f < 4; ++f) {
#pragma unroll
            for (int r = 0; r < 8; ++r) {
                float p = __expf(sc[f][r] - Mrun[r]);
                lsum[r] += p;
                int qq = wave * 16 + hi * 8 + r;
                Ps[qq * A_LD + f * 16 + ln] = f2bf(p);
            }
        }
#pragma unroll
        for (int off = 8; off >= 1; off >>= 1)
#pragma unroll
            for (int r = 0; r < 8; ++r)
                lsum[r] += __shfl_xor(lsum[r], off, 32);
#pragma unroll
        for (int r = 0; r < 8; ++r) Lrun[r] = Lrun[r] * alpha[r] + lsum[r];
        // rescale O accumulators (same row mapping as score C tiles)
#pragma unroll
        for (int f = 0; f < 4; ++f)
#pragma unroll
            for (int r = 0; r < 8; ++r) O[f][r] *= alpha[r];

        __syncthreads();   // P visible (and protects Vs ordering)

        // ---- O += P * V  (key-dim 64 -> 2 WMMA steps; 4 d-blocks of 16)
#pragma unroll
        for (int ks = 0; ks < 2; ++ks) {
            v16bf ap = load_a_frag(Ps + wave * 16 * A_LD + ks * 32, A_LD, lane);
#pragma unroll
            for (int f = 0; f < 4; ++f) {
                v16bf bv = load_b_frag(Vs + f * 16 * A_LD + ks * 32, A_LD, lane);
                O[f] = wmma_bf16(ap, bv, O[f]);
            }
        }
    }

    // ---- normalize and store attention output (bf16, heads re-concatenated)
    bf16_t* Og = AO + ((size_t)b * SEQ + q0) * D_MODEL + h * DKH;
#pragma unroll
    for (int f = 0; f < 4; ++f) {
#pragma unroll
        for (int r = 0; r < 8; ++r) {
            int qq = wave * 16 + hi * 8 + r;
            float o = O[f][r] / Lrun[r];
            Og[(size_t)qq * D_MODEL + f * 16 + ln] = f2bf(o);
        }
    }
}

// ---------------------------------------------------------------------------
// Host launcher
// ---------------------------------------------------------------------------
extern "C" void kernel_launch(void* const* d_in, const int* in_sizes, int n_in,
                              void* d_out, int out_size, void* d_ws, size_t ws_size,
                              hipStream_t stream) {
    const float* q    = (const float*)d_in[0];
    const float* kin  = (const float*)d_in[1];
    const float* vin  = (const float*)d_in[2];
    const int*   mask = (const int*)  d_in[3];
    const float* w_q  = (const float*)d_in[4];
    const float* b_q  = (const float*)d_in[5];
    const float* w_k  = (const float*)d_in[6];
    const float* b_k  = (const float*)d_in[7];
    const float* w_v  = (const float*)d_in[8];
    const float* b_v  = (const float*)d_in[9];
    const float* w_o  = (const float*)d_in[10];
    const float* b_o  = (const float*)d_in[11];
    (void)in_sizes; (void)n_in; (void)out_size; (void)ws_size;

    const size_t nAct = (size_t)BATCH * SEQ * D_MODEL;   // 4,194,304
    const size_t nW   = (size_t)D_MODEL * D_MODEL;       // 1,048,576

    // Workspace carve-out (~42 MB): act buffer reused for q/k/v bf16 and
    // later for the attention output.
    char* base = (char*)d_ws;
    size_t off = 0;
    auto carve = [&](size_t bytes) -> char* {
        char* p = base + off;
        off += (bytes + 255) & ~(size_t)255;
        return p;
    };
    bf16_t* act = (bf16_t*)carve(nAct * 2);
    bf16_t* Qp  = (bf16_t*)carve(nAct * 2);
    bf16_t* Kp  = (bf16_t*)carve(nAct * 2);
    bf16_t* VpT = (bf16_t*)carve(nAct * 2);   // [b][h][d][s]
    bf16_t* wqb = (bf16_t*)carve(nW * 2);     // all weights stored [N][K]
    bf16_t* wkb = (bf16_t*)carve(nW * 2);
    bf16_t* wvb = (bf16_t*)carve(nW * 2);
    bf16_t* wob = (bf16_t*)carve(nW * 2);

    const int BS = BATCH * SEQ;            // 4096 rows
    dim3 blk(128);
    dim3 gGemm(D_MODEL / 64, BS / 64);     // (16, 64)
    dim3 gAtt(SEQ / 64, NHEAD, BATCH);     // (32, 16, 2)
    const int gc  = (int)((nAct + 255) / 256);
    const int gcw = (int)((nW + 255) / 256);

    // Weight conversions: f32 [K][N] -> bf16 [N][K] (transposed once)
    cvt_f32_bf16_T<<<gcw, 256, 0, stream>>>(w_q, wqb, D_MODEL, D_MODEL);
    cvt_f32_bf16_T<<<gcw, 256, 0, stream>>>(w_k, wkb, D_MODEL, D_MODEL);
    cvt_f32_bf16_T<<<gcw, 256, 0, stream>>>(w_v, wvb, D_MODEL, D_MODEL);
    cvt_f32_bf16_T<<<gcw, 256, 0, stream>>>(w_o, wob, D_MODEL, D_MODEL);

    // Q projection
    cvt_f32_bf16<<<gc, 256, 0, stream>>>(q, act, (int)nAct);
    gemm_bf16<<<gGemm, blk, 0, stream>>>(act, wqb, b_q, nullptr, Qp, nullptr,
                                         BS, D_MODEL, D_MODEL);
    // K projection
    cvt_f32_bf16<<<gc, 256, 0, stream>>>(kin, act, (int)nAct);
    gemm_bf16<<<gGemm, blk, 0, stream>>>(act, wkb, b_k, nullptr, Kp, nullptr,
                                         BS, D_MODEL, D_MODEL);
    // V projection -> head-transposed output
    cvt_f32_bf16<<<gc, 256, 0, stream>>>(vin, act, (int)nAct);
    gemm_bf16<<<gGemm, blk, 0, stream>>>(act, wvb, b_v, nullptr, nullptr, VpT,
                                         BS, D_MODEL, D_MODEL);

    // Fused flash attention -> act (reused)
    attention_bf16<<<gAtt, blk, 0, stream>>>(Qp, Kp, VpT, mask, act);

    // Output projection (f32 out + bias)
    gemm_bf16<<<gGemm, blk, 0, stream>>>(act, wob, b_o, (float*)d_out, nullptr,
                                         nullptr, BS, D_MODEL, D_MODEL);
}
